// SelfConvAttentionBlock_54271206752550
// MI455X (gfx1250) — compile-verified, matrix operand order checked by reading, not run on hardware
//
#include <hip/hip_runtime.h>
#include <hip/hip_bf16.h>

// ---------------------------------------------------------------------------
// MI455X (gfx1250) CDNA5 implementation: GroupNorm + QKV + MHSA + proj + res.
// Matmuls via v_wmma_f32_16x16x32_bf16; attention tile staging via
// global_load_async_to_lds_b128 (ASYNCcnt) — both gfx1250-specific paths.
// ---------------------------------------------------------------------------

typedef __bf16 bf16;
typedef __attribute__((ext_vector_type(16))) __bf16 v16bf;
typedef __attribute__((ext_vector_type(8)))  __bf16 v8bf;
typedef __attribute__((ext_vector_type(8)))  float  v8f;

#define B_  16
#define C_  512
#define T_  1024
#define H_  8
#define CH_ 64

struct alignas(16) B16x8 { bf16 h[8]; };
struct alignas(8)  B16x4 { bf16 h[4]; };

__device__ __forceinline__ bf16 f2bf(float x) {
  unsigned u = __float_as_uint(x);
  u += 0x7fffu + ((u >> 16) & 1u);            // round-to-nearest-even
  unsigned short s = (unsigned short)(u >> 16);
  return __builtin_bit_cast(bf16, s);
}

__device__ __forceinline__ v8f zero8() {
  v8f z = {0.f, 0.f, 0.f, 0.f, 0.f, 0.f, 0.f, 0.f};
  return z;
}

// Async 16B-per-lane global -> LDS copy (gfx1250). INST_OFFSET is added to
// both the global and the LDS address, so one base covers chunked copies.
__device__ __forceinline__ void async_cp16(unsigned lds_off, unsigned long long gaddr) {
  asm volatile("global_load_async_to_lds_b128 %0, %1, off"
               :: "v"(lds_off), "v"(gaddr) : "memory");
}
__device__ __forceinline__ void async_cp16_o16(unsigned lds_off, unsigned long long gaddr) {
  asm volatile("global_load_async_to_lds_b128 %0, %1, off offset:16"
               :: "v"(lds_off), "v"(gaddr) : "memory");
}
__device__ __forceinline__ void async_cp16_o32(unsigned lds_off, unsigned long long gaddr) {
  asm volatile("global_load_async_to_lds_b128 %0, %1, off offset:32"
               :: "v"(lds_off), "v"(gaddr) : "memory");
}
__device__ __forceinline__ void async_cp16_o48(unsigned lds_off, unsigned long long gaddr) {
  asm volatile("global_load_async_to_lds_b128 %0, %1, off offset:48"
               :: "v"(lds_off), "v"(gaddr) : "memory");
}
__device__ __forceinline__ void wait_async0() {
  asm volatile("s_wait_asynccnt 0x0" ::: "memory");
}

// Load one 16x32 bf16 WMMA operand fragment from an LDS tile stored so that
// the fragment's 16 K-values per lane are two contiguous 8-element runs:
//   A operand : tile is [M][ld] row-major       (row = M)
//   B operand : tile is [N][ld] ("transposed")  (row = N)
// ISA 16-bit layout: lanes 0-15 -> K 0-7 & 16-23, lanes 16-31 -> K 8-15 & 24-31.
__device__ __forceinline__ v16bf load_frag(const bf16* base, int row, int k0, int ld) {
  const int lane = threadIdx.x & 31;
  const bf16* p = base + (size_t)(row + (lane & 15)) * ld + k0 + ((lane >> 4) << 3);
  v8bf lo = *(const v8bf*)p;
  v8bf hi = *(const v8bf*)(p + 16);
  v16bf a;
#pragma unroll
  for (int i = 0; i < 8; ++i) { a[i] = lo[i]; a[i + 8] = hi[i]; }
  return a;
}

__device__ __forceinline__ v8f wmma_bf16(v16bf a, v16bf b, v8f c) {
  return __builtin_amdgcn_wmma_f32_16x16x32_bf16(false, a, false, b,
                                                 (short)0, c, false, false);
}

// ---------------------------------------------------------------------------
// K1: GroupNorm statistics. One block per (b, group): 16 ch * 1024 contiguous.
// ---------------------------------------------------------------------------
__global__ __launch_bounds__(256) void gn_stats(const float* __restrict__ x,
                                                float* __restrict__ stats) {
  __shared__ float rs[256], rq[256];
  const int bg = blockIdx.x;                       // b*32 + g
  const float* p = x + (size_t)bg * 16384;
  float s = 0.f, q = 0.f;
  for (int i = threadIdx.x; i < 16384; i += 256) { float v = p[i]; s += v; q += v * v; }
  rs[threadIdx.x] = s; rq[threadIdx.x] = q;
  __syncthreads();
  for (int o = 128; o > 0; o >>= 1) {
    if ((int)threadIdx.x < o) { rs[threadIdx.x] += rs[threadIdx.x + o];
                                rq[threadIdx.x] += rq[threadIdx.x + o]; }
    __syncthreads();
  }
  if (threadIdx.x == 0) {
    float mean = rs[0] * (1.f / 16384.f);
    float var  = rq[0] * (1.f / 16384.f) - mean * mean;
    stats[bg * 2]     = mean;
    stats[bg * 2 + 1] = rsqrtf(var + 1e-5f);
  }
}

// ---------------------------------------------------------------------------
// K2: normalize + affine, emit bf16 xn[B][C][T].
// ---------------------------------------------------------------------------
__global__ __launch_bounds__(256) void gn_apply(const float* __restrict__ x,
                                                const float* __restrict__ stats,
                                                const float* __restrict__ gw,
                                                const float* __restrict__ gb,
                                                bf16* __restrict__ xn) {
  const size_t base = ((size_t)blockIdx.x * blockDim.x + threadIdx.x) * 4;  // 4 elems, same (b,c)
  const int c = (int)((base >> 10) & 511);
  const int b = (int)(base >> 19);
  const int g = c >> 4;
  const float mean = stats[((b << 5) + g) * 2];
  const float rstd = stats[((b << 5) + g) * 2 + 1];
  const float sw = gw[c] * rstd;
  const float sb = gb[c] - mean * sw;
  float4 v = *(const float4*)(x + base);
  B16x4 o;
  o.h[0] = f2bf(v.x * sw + sb);
  o.h[1] = f2bf(v.y * sw + sb);
  o.h[2] = f2bf(v.z * sw + sb);
  o.h[3] = f2bf(v.w * sw + sb);
  *(B16x4*)(xn + base) = o;
}

// ---------------------------------------------------------------------------
// K3: f32 -> bf16 weight conversion.
// ---------------------------------------------------------------------------
__global__ __launch_bounds__(256) void cvt_bf16(const float* __restrict__ s,
                                                bf16* __restrict__ d, int n) {
  int i = blockIdx.x * blockDim.x + threadIdx.x;
  if (i < n) d[i] = f2bf(s[i]);
}

// ---------------------------------------------------------------------------
// K4: QKV GEMM.  qkv[o,t] = sum_c W[o,c] * xn[c,t]  (per batch)
//   M = 1536 (o), N = 1024 (t), K = 512 (c). Block tile 128x128, K-step 32.
//   Waves: 4 (M) x 2 (N); each wave: 2x4 tiles of 16x16 -> 8 WMMAs / K-step.
//   Epilogue: Q -> [b,h,t,c]*scale, K -> [b,h,s,c]*scale, V -> [b,h,c,s].
// ---------------------------------------------------------------------------
__global__ __launch_bounds__(256) void qkv_gemm(const bf16* __restrict__ W,
                                                const bf16* __restrict__ Xn,
                                                const float* __restrict__ bias,
                                                bf16* __restrict__ Q,
                                                bf16* __restrict__ K,
                                                bf16* __restrict__ V) {
  __shared__ __align__(16) bf16 As[128 * 32];   // [m][k]
  __shared__ __align__(16) bf16 Bs[128 * 32];   // [n][k]  (transposed stage)
  const int tid = threadIdx.x;
  const int lane = tid & 31, wid = tid >> 5, hi = lane >> 4;
  const int wm = wid & 3, wn = wid >> 2;
  const int b  = blockIdx.z;
  const int m0 = blockIdx.y * 128;
  const int n0 = blockIdx.x * 128;

  v8f acc[2][4];
#pragma unroll
  for (int i = 0; i < 2; ++i)
#pragma unroll
    for (int j = 0; j < 4; ++j) acc[i][j] = zero8();

  for (int kb = 0; kb < 16; ++kb) {
    { // stage A: W rows, contiguous
      int row = tid >> 1, half = tid & 1;
      const bf16* src = W + (size_t)(m0 + row) * 512 + kb * 32 + half * 16;
      __builtin_prefetch(src + 32, 0, 1);                 // -> global_prefetch_b8
      *(B16x8*)&As[row * 32 + half * 16]     = *(const B16x8*)src;
      *(B16x8*)&As[row * 32 + half * 16 + 8] = *(const B16x8*)(src + 8);
    }
    { // stage B transposed: Bs[t][c]
      int c = tid & 31, tt0 = (tid >> 5) * 16;
      const bf16* src = Xn + ((size_t)b * 512 + kb * 32 + c) * 1024 + n0 + tt0;
      B16x8 t0 = *(const B16x8*)src;
      B16x8 t1 = *(const B16x8*)(src + 8);
#pragma unroll
      for (int i = 0; i < 8; ++i) Bs[(tt0 + i) * 32 + c] = t0.h[i];
#pragma unroll
      for (int i = 0; i < 8; ++i) Bs[(tt0 + 8 + i) * 32 + c] = t1.h[i];
    }
    __syncthreads();
    v16bf af[2], bfm[4];
#pragma unroll
    for (int tm = 0; tm < 2; ++tm) af[tm] = load_frag(As, wm * 32 + tm * 16, 0, 32);
#pragma unroll
    for (int tn = 0; tn < 4; ++tn) bfm[tn] = load_frag(Bs, wn * 64 + tn * 16, 0, 32);
#pragma unroll
    for (int tm = 0; tm < 2; ++tm)
#pragma unroll
      for (int tn = 0; tn < 4; ++tn)
        acc[tm][tn] = wmma_bf16(af[tm], bfm[tn], acc[tm][tn]);
    __syncthreads();
  }

  const float scale = 0.3535533906f;                      // 1 / 64^0.25
#pragma unroll
  for (int tm = 0; tm < 2; ++tm) {
    const int m_base = m0 + wm * 32 + tm * 16 + hi * 8;   // 8 consecutive rows/lane
#pragma unroll
    for (int tn = 0; tn < 4; ++tn) {
      const int n_g = n0 + wn * 64 + tn * 16 + (lane & 15);
      B16x8 pk;
      if (m_base < 512) {                                 // Q -> [b,h,t,c]
        const int h = m_base >> 6, c0 = m_base & 63;
#pragma unroll
        for (int r = 0; r < 8; ++r)
          pk.h[r] = f2bf((acc[tm][tn][r] + bias[m_base + r]) * scale);
        *(B16x8*)&Q[(((size_t)b * H_ + h) * T_ + n_g) * CH_ + c0] = pk;
      } else if (m_base < 1024) {                         // K -> [b,h,s,c]
        const int mk = m_base - 512, h = mk >> 6, c0 = mk & 63;
#pragma unroll
        for (int r = 0; r < 8; ++r)
          pk.h[r] = f2bf((acc[tm][tn][r] + bias[m_base + r]) * scale);
        *(B16x8*)&K[(((size_t)b * H_ + h) * T_ + n_g) * CH_ + c0] = pk;
      } else {                                            // V -> [b,h,c,s]
        const int mv = m_base - 1024, h = mv >> 6, c0 = mv & 63;
#pragma unroll
        for (int r = 0; r < 8; ++r)
          V[(((size_t)b * H_ + h) * CH_ + c0 + r) * (size_t)T_ + n_g] =
              f2bf(acc[tm][tn][r] + bias[m_base + r]);
      }
    }
  }
}

// ---------------------------------------------------------------------------
// K5: fused flash attention. One WG per (b, h, 128 q-rows); wave owns 16 rows.
//   s-step = 64. GEMM1: S=Q*K^T (8 WMMAs), online softmax, GEMM2: O+=P*V (8).
//   Q/K/V tiles staged with global_load_async_to_lds_b128 (ASYNCcnt).
// ---------------------------------------------------------------------------
__global__ __launch_bounds__(256) void attn(const bf16* __restrict__ Q,
                                            const bf16* __restrict__ K,
                                            const bf16* __restrict__ V,
                                            bf16* __restrict__ A) {
  __shared__ __align__(16) bf16 Qs[128 * 64];     // [t][c]
  __shared__ __align__(16) bf16 Ks[64 * 64];      // [s][c]
  __shared__ __align__(16) bf16 Vs[64 * 64];      // [c][s]
  __shared__ __align__(16) bf16 Ps[8][16 * 64];   // per-wave P in A-frag layout
  const int tid = threadIdx.x, lane = tid & 31, wid = tid >> 5, hi = lane >> 4;
  const int b = blockIdx.z, h = blockIdx.y, qt0 = blockIdx.x * 128;
  const size_t bh = (size_t)b * H_ + h;

  { // stage Q tile (128 x 64): 64B per thread, async DMA to LDS
    int row = tid >> 1, half = tid & 1;
    const bf16* src = Q + (bh * T_ + qt0 + row) * CH_ + half * 32;
    unsigned dst = (unsigned)(uintptr_t)&Qs[row * 64 + half * 32];
    unsigned long long ga = (unsigned long long)(uintptr_t)src;
    async_cp16(dst, ga);
    async_cp16_o16(dst, ga);
    async_cp16_o32(dst, ga);
    async_cp16_o48(dst, ga);
  }

  v8f O[4];
#pragma unroll
  for (int tc = 0; tc < 4; ++tc) O[tc] = zero8();
  float mrun[8], lrun[8];
#pragma unroll
  for (int r = 0; r < 8; ++r) { mrun[r] = -3.0e38f; lrun[r] = 0.f; }

  for (int st = 0; st < 16; ++st) {
    const int s0 = st * 64;
    { // stage K (s,c) and V (c,s) tiles: straight copies -> async DMA
      int row = tid >> 2, seg = tid & 3;
      const bf16* ksrc = K + (bh * T_ + s0 + row) * CH_ + seg * 16;
      unsigned kdst = (unsigned)(uintptr_t)&Ks[row * 64 + seg * 16];
      unsigned long long ka = (unsigned long long)(uintptr_t)ksrc;
      async_cp16(kdst, ka);
      async_cp16_o16(kdst, ka);
      const bf16* vsrc = V + (bh * CH_ + row) * (size_t)T_ + s0 + seg * 16;
      unsigned vdst = (unsigned)(uintptr_t)&Vs[row * 64 + seg * 16];
      unsigned long long va = (unsigned long long)(uintptr_t)vsrc;
      async_cp16(vdst, va);
      async_cp16_o16(vdst, va);
    }
    wait_async0();           // s_wait_asynccnt 0 (covers Q on first iteration)
    __syncthreads();

    // ---- GEMM1: S[16 x 64] = Q_tile * K_tile^T (contraction over c=64)
    v8f S[4];
#pragma unroll
    for (int tn = 0; tn < 4; ++tn) S[tn] = zero8();
    v16bf qa[2];
#pragma unroll
    for (int ks = 0; ks < 2; ++ks) qa[ks] = load_frag(Qs, wid * 16, ks * 32, 64);
#pragma unroll
    for (int tn = 0; tn < 4; ++tn)
#pragma unroll
      for (int ks = 0; ks < 2; ++ks)
        S[tn] = wmma_bf16(qa[ks], load_frag(Ks, tn * 16, ks * 32, 64), S[tn]);

    // ---- online softmax: row r lives at vgpr r (+8 for lanes 16-31)
#pragma unroll
    for (int r = 0; r < 8; ++r) {
      float mx = S[0][r];
#pragma unroll
      for (int tn = 1; tn < 4; ++tn) mx = fmaxf(mx, S[tn][r]);
#pragma unroll
      for (int off = 1; off < 16; off <<= 1) mx = fmaxf(mx, __shfl_xor(mx, off, 32));
      const float mnew  = fmaxf(mrun[r], mx);
      const float alpha = __expf(mrun[r] - mnew);
      mrun[r] = mnew;
      float rsum = 0.f;
#pragma unroll
      for (int tn = 0; tn < 4; ++tn) {
        float p = __expf(S[tn][r] - mnew);
        S[tn][r] = p;
        rsum += p;
      }
#pragma unroll
      for (int off = 1; off < 16; off <<= 1) rsum += __shfl_xor(rsum, off, 32);
      lrun[r] = lrun[r] * alpha + rsum;
#pragma unroll
      for (int tc = 0; tc < 4; ++tc) O[tc][r] *= alpha;
    }

    // ---- P (bf16) -> per-wave LDS patch in A-fragment layout [m][s]
#pragma unroll
    for (int tn = 0; tn < 4; ++tn)
#pragma unroll
      for (int r = 0; r < 8; ++r)
        Ps[wid][(r + hi * 8) * 64 + tn * 16 + (lane & 15)] = f2bf(S[tn][r]);

    // ---- GEMM2: O[16 x 64] += P * V_tile (contraction over s=64)
    v16bf pa[2];
#pragma unroll
    for (int ks = 0; ks < 2; ++ks) pa[ks] = load_frag(Ps[wid], 0, ks * 32, 64);
#pragma unroll
    for (int tc = 0; tc < 4; ++tc)
#pragma unroll
      for (int ks = 0; ks < 2; ++ks)
        O[tc] = wmma_bf16(pa[ks], load_frag(Vs, tc * 16, ks * 32, 64), O[tc]);
    __syncthreads();
  }

  // ---- epilogue: normalize rows, store a[b][c][t] (bf16) for proj GEMM
#pragma unroll
  for (int tc = 0; tc < 4; ++tc) {
    const int cg = h * CH_ + tc * 16 + (lane & 15);
    const int t0 = qt0 + wid * 16 + hi * 8;
    B16x8 pk;
#pragma unroll
    for (int r = 0; r < 8; ++r) pk.h[r] = f2bf(O[tc][r] * (1.0f / lrun[r]));
    *(B16x8*)&A[((size_t)b * C_ + cg) * T_ + t0] = pk;
  }
}

// ---------------------------------------------------------------------------
// K6: projection GEMM + bias + residual.  out = x + Wp * a + pb
//   M = 512 (o), N = 1024 (t), K = 512 (c). Same tiling as K4.
// ---------------------------------------------------------------------------
__global__ __launch_bounds__(256) void proj_gemm(const bf16* __restrict__ W,
                                                 const bf16* __restrict__ Ain,
                                                 const float* __restrict__ bias,
                                                 const float* __restrict__ X,
                                                 float* __restrict__ Out) {
  __shared__ __align__(16) bf16 As[128 * 32];
  __shared__ __align__(16) bf16 Bs[128 * 32];
  const int tid = threadIdx.x;
  const int lane = tid & 31, wid = tid >> 5, hi = lane >> 4;
  const int wm = wid & 3, wn = wid >> 2;
  const int b  = blockIdx.z;
  const int m0 = blockIdx.y * 128;
  const int n0 = blockIdx.x * 128;

  v8f acc[2][4];
#pragma unroll
  for (int i = 0; i < 2; ++i)
#pragma unroll
    for (int j = 0; j < 4; ++j) acc[i][j] = zero8();

  for (int kb = 0; kb < 16; ++kb) {
    {
      int row = tid >> 1, half = tid & 1;
      const bf16* src = W + (size_t)(m0 + row) * 512 + kb * 32 + half * 16;
      *(B16x8*)&As[row * 32 + half * 16]     = *(const B16x8*)src;
      *(B16x8*)&As[row * 32 + half * 16 + 8] = *(const B16x8*)(src + 8);
    }
    {
      int c = tid & 31, tt0 = (tid >> 5) * 16;
      const bf16* src = Ain + ((size_t)b * 512 + kb * 32 + c) * 1024 + n0 + tt0;
      __builtin_prefetch(src + 32768, 0, 1);
      B16x8 t0 = *(const B16x8*)src;
      B16x8 t1 = *(const B16x8*)(src + 8);
#pragma unroll
      for (int i = 0; i < 8; ++i) Bs[(tt0 + i) * 32 + c] = t0.h[i];
#pragma unroll
      for (int i = 0; i < 8; ++i) Bs[(tt0 + 8 + i) * 32 + c] = t1.h[i];
    }
    __syncthreads();
    v16bf af[2], bfm[4];
#pragma unroll
    for (int tm = 0; tm < 2; ++tm) af[tm] = load_frag(As, wm * 32 + tm * 16, 0, 32);
#pragma unroll
    for (int tn = 0; tn < 4; ++tn) bfm[tn] = load_frag(Bs, wn * 64 + tn * 16, 0, 32);
#pragma unroll
    for (int tm = 0; tm < 2; ++tm)
#pragma unroll
      for (int tn = 0; tn < 4; ++tn)
        acc[tm][tn] = wmma_bf16(af[tm], bfm[tn], acc[tm][tn]);
    __syncthreads();
  }

#pragma unroll
  for (int tm = 0; tm < 2; ++tm) {
    const int o_base = m0 + wm * 32 + tm * 16 + hi * 8;
#pragma unroll
    for (int tn = 0; tn < 4; ++tn) {
      const int t = n0 + wn * 64 + tn * 16 + (lane & 15);
#pragma unroll
      for (int r = 0; r < 8; ++r) {
        const int o = o_base + r;
        const size_t idx = ((size_t)b * C_ + o) * T_ + t;
        Out[idx] = X[idx] + bias[o] + acc[tm][tn][r];
      }
    }
  }
}

// ---------------------------------------------------------------------------
// Host launcher
// ---------------------------------------------------------------------------
extern "C" void kernel_launch(void* const* d_in, const int* in_sizes, int n_in,
                              void* d_out, int out_size, void* d_ws, size_t ws_size,
                              hipStream_t stream) {
  (void)in_sizes; (void)n_in; (void)out_size; (void)ws_size;
  const float* x      = (const float*)d_in[0];
  // d_in[1] = emb : unused by the reference
  const float* gn_w   = (const float*)d_in[2];
  const float* gn_b   = (const float*)d_in[3];
  const float* qkv_w  = (const float*)d_in[4];
  const float* qkv_b  = (const float*)d_in[5];
  const float* proj_w = (const float*)d_in[6];
  const float* proj_b = (const float*)d_in[7];
  float* out = (float*)d_out;

  char* ws = (char*)d_ws;
  size_t off = 0;
  auto carve = [&](size_t bytes) -> void* {
    void* p = ws + off;
    off = (off + bytes + 255) & ~(size_t)255;
    return p;
  };
  float* stats = (float*)carve((size_t)B_ * 32 * 2 * sizeof(float));
  bf16* xn     = (bf16*)carve((size_t)B_ * C_ * T_ * sizeof(bf16));
  bf16* wqkv   = (bf16*)carve((size_t)3 * C_ * C_ * sizeof(bf16));
  bf16* wproj  = (bf16*)carve((size_t)C_ * C_ * sizeof(bf16));
  bf16* qb     = (bf16*)carve((size_t)B_ * H_ * T_ * CH_ * sizeof(bf16));
  bf16* kb     = (bf16*)carve((size_t)B_ * H_ * T_ * CH_ * sizeof(bf16));
  bf16* vb     = (bf16*)carve((size_t)B_ * H_ * T_ * CH_ * sizeof(bf16));
  bf16* abuf   = (bf16*)carve((size_t)B_ * C_ * T_ * sizeof(bf16));

  gn_stats<<<B_ * 32, 256, 0, stream>>>(x, stats);
  gn_apply<<<(B_ * C_ * T_) / (256 * 4), 256, 0, stream>>>(x, stats, gn_w, gn_b, xn);
  cvt_bf16<<<(3 * C_ * C_) / 256, 256, 0, stream>>>(qkv_w, wqkv, 3 * C_ * C_);
  cvt_bf16<<<(C_ * C_) / 256, 256, 0, stream>>>(proj_w, wproj, C_ * C_);
  qkv_gemm<<<dim3(T_ / 128, (3 * C_) / 128, B_), 256, 0, stream>>>(wqkv, xn, qkv_b,
                                                                   qb, kb, vb);
  attn<<<dim3(T_ / 128, H_, B_), 256, 0, stream>>>(qb, kb, vb, abuf);
  proj_gemm<<<dim3(T_ / 128, C_ / 128, B_), 256, 0, stream>>>(wproj, abuf, proj_b,
                                                              x, out);
}
